// Network_56684978372986
// MI455X (gfx1250) — compile-verified, matrix operand order checked by reading, not run on hardware
//
#include <hip/hip_runtime.h>
#include <hip/hip_bf16.h>

typedef __attribute__((ext_vector_type(16))) _Float16 v16h;
typedef __attribute__((ext_vector_type(8)))  _Float16 v8h;
typedef __attribute__((ext_vector_type(8)))  float    v8f;

#define T_STEPS 300
#define K_PSP   100
#define THETA   1.0f

__device__ __forceinline__ int imin(int a, int b) { return a < b ? a : b; }
__device__ __forceinline__ int imax(int a, int b) { return a > b ? a : b; }

// ---------------------------------------------------------------------------
// One-time f32 -> f16 conversion (dense weights), done in workspace each call.
// ---------------------------------------------------------------------------
__global__ void f32_to_f16_kernel(const float* __restrict__ in,
                                  _Float16* __restrict__ out, int n)
{
    const int i = blockIdx.x * blockDim.x + threadIdx.x;
    if (i < n) out[i] = (_Float16)in[i];
}

// ---------------------------------------------------------------------------
// Fused per-row (n,c,w):  optional per-channel fractional delay  ->  psp filter
// One block per row (row = ((n*C)+c)*Wsp + w); whole T=300 series in LDS.
// dly == nullptr  -> delay stage is identity (first psp on raw spikes).
// transposed == 0 -> f32 out[row*T + t]                        (safe in-place)
// transposed == 1 -> f16 ((_Float16*)out)[((n*T)+t)*Fdim + f]  (out-of-place),
//                    f = c*Wsp + w  — feeds the WMMA B-matrix contiguously in K.
// ---------------------------------------------------------------------------
__global__ void delay_psp_kernel(const float* __restrict__ in,
                                 float* __restrict__ out,
                                 const float* __restrict__ dly,
                                 int C, int Wsp, int T,
                                 int transposed, int Fdim)
{
    __shared__ float sx[T_STEPS];
    __shared__ float sd[T_STEPS];
    __shared__ float seps[K_PSP];

    const int r = blockIdx.x;
    const int c = (r / Wsp) % C;
    const size_t base = (size_t)r * T;

    for (int t = threadIdx.x; t < T; t += blockDim.x) sx[t] = in[base + t];
    if (threadIdx.x < K_PSP) {
        float k = (float)threadIdx.x;
        // eps(k) = (k/tau) * exp(1 - k/tau), tau = 10
        seps[threadIdx.x] = (k * 0.1f) * __expf(1.0f - k * 0.1f);
    }
    __syncthreads();

    if (dly != nullptr) {
        const float dd = dly[c];
        const int   tf = (int)floorf(dd);
        const float fr = dd - (float)tf;
        for (int t = threadIdx.x; t < T; t += blockDim.x) {
            int i0 = t - tf, i1 = i0 - 1;
            float g0 = (i0 >= 0) ? sx[imin(imax(i0, 0), T - 1)] : 0.0f;
            float g1 = (i1 >= 0) ? sx[imin(imax(i1, 0), T - 1)] : 0.0f;
            sd[t] = (1.0f - fr) * g0 + fr * g1;
        }
    } else {
        for (int t = threadIdx.x; t < T; t += blockDim.x) sd[t] = sx[t];
    }
    __syncthreads();

    const int fidx = r % (C * Wsp);
    const int nidx = r / (C * Wsp);
    for (int t = threadIdx.x; t < T; t += blockDim.x) {
        float s = 0.0f;
        const int kmax = imin(t, K_PSP - 1);
        for (int k = 0; k <= kmax; ++k) s += seps[k] * sd[t - k];
        if (transposed)
            ((_Float16*)out)[((size_t)nidx * T + t) * Fdim + fidx] = (_Float16)s;
        else
            out[base + t] = s;
    }
}

// ---------------------------------------------------------------------------
// Per-timestep 1x3 conv (stride 2 in W) + spike threshold.
// in : [N][Cin][Win][T]   w : [Cout][Cin][1][3]   out : [N][Cout][Wout][T]
// ---------------------------------------------------------------------------
__global__ void conv_spike_kernel(const float* __restrict__ in,
                                  const float* __restrict__ w,
                                  float* __restrict__ out,
                                  int N, int Cin, int Win,
                                  int Cout, int Wout, int padl, int T)
{
    const long long idx = (long long)blockIdx.x * blockDim.x + threadIdx.x;
    const long long total = (long long)N * Cout * Wout * T;
    if (idx >= total) return;

    const int t  = (int)(idx % T);
    long long r  = idx / T;
    const int wo = (int)(r % Wout); r /= Wout;
    const int co = (int)(r % Cout);
    const int n  = (int)(r / Cout);

    float acc = 0.0f;
    const int wbase = co * Cin * 3;
    for (int ci = 0; ci < Cin; ++ci) {
        const float* irow = in + ((size_t)(n * Cin + ci) * Win) * T + t;
        for (int kx = 0; kx < 3; ++kx) {
            const int wi = wo * 2 - padl + kx;
            if (wi >= 0 && wi < Win)
                acc += w[wbase + ci * 3 + kx] * irow[(size_t)wi * T];
        }
    }
    out[idx] = (acc >= THETA) ? 1.0f : 0.0f;
}

// ---------------------------------------------------------------------------
// Dense per-timestep layer as WMMA GEMM + spike epilogue (all-f16 operands).
// XT  : [N][T][F]  f16 transposed psp activations
// Wh  : [O][F]     f16 weights
// Out : [N][O][T]  f32 spike output
// 256-thread block = 8 waves; each wave owns one 16(o) x 16(t) tile of batch n.
// Out-of-range rows/cols load CLAMPED (valid) addresses unconditionally; junk
// lands only in D entries masked at the store -> no exec churn in the hot loop.
// Per WMMA: 2 b128 loads for A + 2 for B, zero conversions.
// ---------------------------------------------------------------------------
__global__ __launch_bounds__(256)
void dense_wmma_spike_kernel(const _Float16* __restrict__ XT,
                             const _Float16* __restrict__ Wh,
                             float* __restrict__ Out,
                             int F, int O, int T)
{
    const int wave = threadIdx.x >> 5;
    const int lane = threadIdx.x & 31;
    const int row  = lane & 15;
    const int half = lane >> 4;
    const int ot0  = (blockIdx.x * 8 + wave) * 16;   // output-channel tile base
    const int tt0  = blockIdx.y * 16;                // time tile base
    const int n    = blockIdx.z;
    if (ot0 >= O) return;                            // wave-uniform exit, no barriers

    const int m    = imin(ot0 + row, O - 1);         // clamped A row
    const int tcol = imin(tt0 + row, T - 1);         // clamped B column (time)

    // ISA 16-bit fragment layout: element e -> K = (e>>3)*16 + half*8 + (e&7)
    // -> per lane, two contiguous 8-half (16B) runs at +half*8 and +16+half*8.
    const _Float16* wrow = Wh + (size_t)m * F + (half << 3);
    const _Float16* xrow = XT + ((size_t)n * T + tcol) * F + (half << 3);

    v8f acc = {};
#pragma unroll 4
    for (int k0 = 0; k0 < F; k0 += 32) {
        const v8h a_lo = *(const v8h*)(wrow + k0);        // K = half*8 + 0..7
        const v8h a_hi = *(const v8h*)(wrow + k0 + 16);   // K = 16 + half*8 + 0..7
        const v8h b_lo = *(const v8h*)(xrow + k0);
        const v8h b_hi = *(const v8h*)(xrow + k0 + 16);

        const v16h a = __builtin_shufflevector(a_lo, a_hi,
            0, 1, 2, 3, 4, 5, 6, 7, 8, 9, 10, 11, 12, 13, 14, 15);
        const v16h b = __builtin_shufflevector(b_lo, b_hi,
            0, 1, 2, 3, 4, 5, 6, 7, 8, 9, 10, 11, 12, 13, 14, 15);

        acc = __builtin_amdgcn_wmma_f32_16x16x32_f16(
            /*neg_a=*/false, a, /*neg_b=*/false, b,
            /*c_mod=*/(short)0, acc, /*reuse_a=*/false, /*reuse_b=*/false);
    }

    const int colt = tt0 + (lane & 15);
#pragma unroll
    for (int rr = 0; rr < 8; ++rr) {
        const int mm = ot0 + half * 8 + rr;   // ISA C/D layout: M = rr + half*8
        if (mm < O && colt < T) {
            const float u = acc[rr];
            Out[((size_t)n * O + mm) * T + colt] = (u >= THETA) ? 1.0f : 0.0f;
        }
    }
}

// ---------------------------------------------------------------------------
// Host-side pipeline
// ---------------------------------------------------------------------------
extern "C" void kernel_launch(void* const* d_in, const int* in_sizes, int n_in,
                              void* d_out, int out_size, void* d_ws, size_t ws_size,
                              hipStream_t stream)
{
    (void)in_sizes; (void)n_in; (void)out_size;

    const float* spike = (const float*)d_in[0];   // (32,1,1,691,300)
    const float* W1    = (const float*)d_in[1];   // (8,1,1,3)
    const float* W2    = (const float*)d_in[2];   // (16,8,1,3)
    const float* W3    = (const float*)d_in[3];   // (32,16,1,3)
    const float* W4    = (const float*)d_in[4];   // (64,32,1,3)
    const float* Wf1   = (const float*)d_in[5];   // (512,2816)
    const float* Wf2   = (const float*)d_in[6];   // (35,512)
    const float* d1    = (const float*)d_in[7];
    const float* d2    = (const float*)d_in[8];
    const float* d3    = (const float*)d_in[9];
    const float* d4    = (const float*)d_in[10];
    const float* d5    = (const float*)d_in[11];

    const int N = 32, T = T_STEPS;
    // widths: 691 -> 345 -> 173 -> 87 -> 44
    const size_t BUF    = (size_t)N * 64 * 44 * T;   // 27,033,600 floats (largest tensor)
    const size_t NWF1   = (size_t)512 * 2816;        // 1,441,792
    const size_t NWF2   = (size_t)35 * 512;          // 17,920
    const size_t needed = 2 * BUF * sizeof(float) + (NWF1 + NWF2) * sizeof(_Float16);
    if (ws_size < needed) return;

    float*     A    = (float*)d_ws;
    float*     B    = A + BUF;
    _Float16*  Wf1h = (_Float16*)(B + BUF);
    _Float16*  Wf2h = Wf1h + NWF1;
    float*     out  = (float*)d_out;

    const int DP_THREADS = 128, CV_THREADS = 256;
    auto cdiv = [](long long a, long long b) { return (int)((a + b - 1) / b); };

    // ---- dense weights -> f16 (once per call, in workspace) ----
    f32_to_f16_kernel<<<cdiv(NWF1, 256), 256, 0, stream>>>(Wf1, Wf1h, (int)NWF1);
    f32_to_f16_kernel<<<cdiv(NWF2, 256), 256, 0, stream>>>(Wf2, Wf2h, (int)NWF2);

    // ---- layer 1 ----
    delay_psp_kernel<<<N * 1 * 691, DP_THREADS, 0, stream>>>(spike, A, nullptr, 1, 691, T, 0, 0);
    conv_spike_kernel<<<cdiv((long long)N * 8 * 345 * T, CV_THREADS), CV_THREADS, 0, stream>>>(
        A, W1, B, N, 1, 691, 8, 345, /*padl=*/0, T);
    delay_psp_kernel<<<N * 8 * 345, DP_THREADS, 0, stream>>>(B, B, d1, 8, 345, T, 0, 0);

    // ---- layer 2 ----
    conv_spike_kernel<<<cdiv((long long)N * 16 * 173 * T, CV_THREADS), CV_THREADS, 0, stream>>>(
        B, W2, A, N, 8, 345, 16, 173, /*padl=*/1, T);
    delay_psp_kernel<<<N * 16 * 173, DP_THREADS, 0, stream>>>(A, A, d2, 16, 173, T, 0, 0);

    // ---- layer 3 ----
    conv_spike_kernel<<<cdiv((long long)N * 32 * 87 * T, CV_THREADS), CV_THREADS, 0, stream>>>(
        A, W3, B, N, 16, 173, 32, 87, /*padl=*/1, T);
    delay_psp_kernel<<<N * 32 * 87, DP_THREADS, 0, stream>>>(B, B, d3, 32, 87, T, 0, 0);

    // ---- layer 4 ----
    conv_spike_kernel<<<cdiv((long long)N * 64 * 44 * T, CV_THREADS), CV_THREADS, 0, stream>>>(
        B, W4, A, N, 32, 87, 64, 44, /*padl=*/1, T);
    // delay4 + psp, written TRANSPOSED as f16 [N][T][2816] into B
    delay_psp_kernel<<<N * 64 * 44, DP_THREADS, 0, stream>>>(A, B, d4, 64, 44, T, 1, 2816);

    // ---- dense 1 : 512 x 2816 WMMA GEMM + spike -> A as f32 [N][512][T] ----
    {
        dim3 grid(512 / 128, (T + 15) / 16, N);   // 8 waves/block cover 128 output rows
        dense_wmma_spike_kernel<<<grid, 256, 0, stream>>>(
            (const _Float16*)B, Wf1h, A, 2816, 512, T);
    }
    // delay5 + psp, TRANSPOSED as f16 [N][T][512] into B
    delay_psp_kernel<<<N * 512, DP_THREADS, 0, stream>>>(A, B, d5, 512, 1, T, 1, 512);

    // ---- dense 2 : 35 x 512 WMMA GEMM + spike -> d_out [N][35][T] ----
    {
        dim3 grid(1, (T + 15) / 16, N);
        dense_wmma_spike_kernel<<<grid, 256, 0, stream>>>(
            (const _Float16*)B, Wf2h, out, 512, 35, T);
    }
}
